// NSE_52140902973440
// MI455X (gfx1250) — matrix operand order repaired and checked by reading, not compiled, not run on hardware
//
#include <hip/hip_runtime.h>
#include <hip/hip_bf16.h>
#include <math.h>

#define B_  8
#define L_  512
#define D_  256
#define D4_ 1024
#define KT_ 16   // K tiles: K=512 / 32

typedef __attribute__((ext_vector_type(16))) __bf16 bf16x16;
typedef __attribute__((ext_vector_type(8)))  float  f32x8;

union BF2   { unsigned u; unsigned short s[2]; __bf16 h[2]; };
union BFRAG { uint4 q[2]; bf16x16 v; };

// manual round-to-nearest-even fp32 -> bf16
__device__ inline unsigned short f2bf(float f) {
  unsigned u = __float_as_uint(f);
  unsigned r = u + 0x7FFFu + ((u >> 16) & 1u);
  return (unsigned short)(r >> 16);
}
__device__ inline unsigned pack_bf16(float a, float b) {
  BF2 c; c.s[0] = f2bf(a); c.s[1] = f2bf(b); return c.u;
}

// Swizzle a dword index into A-fragment order: within each ktile's 16 dwords,
// swap blocks [4..7] <-> [8..11]. After this, the 8 dwords each lane needs for
// an A fragment are contiguous: lanes 0-15 read kt*16+0..7, lanes 16-31 read
// kt*16+8..15  ->  exactly two aligned ds_load_b128 per fragment.
__device__ inline int swz(int w) {
  int r = w & 15;
  int f = ((r >> 2) ^ (r >> 3)) & 1;
  return (w & ~15) | (r ^ (f * 12));
}

// A fragment (16x32 bf16, ISA 7.12.2 layout) from pre-swizzled LDS.
// M=1 trick: all lanes load the same row-0 data (LDS broadcast); accumulator
// rows 1..15 are garbage we never read (rows don't mix in a matmul).
__device__ inline bf16x16 load_a_frag(const unsigned* xh_dw, int ktile, int lane) {
  const int half = lane >> 4;
  const uint4* p = (const uint4*)(xh_dw + ktile * 16 + half * 8);
  BFRAG f; f.q[0] = p[0]; f.q[1] = p[1];
  return f.v;
}

// B fragment (32x16 bf16): pre-packed tiles, each lane reads 32 contiguous bytes.
__device__ inline bf16x16 load_b_frag(const unsigned* wtiles, int tile, int lane) {
  const uint4* p = (const uint4*)(wtiles + ((size_t)tile * 32 + lane) * 8);
  BFRAG f; f.q[0] = p[0]; f.q[1] = p[1];
  return f.v;
}

__device__ inline f32x8 wmma_bf16(bf16x16 a, bf16x16 b, f32x8 c) {
  return __builtin_amdgcn_wmma_f32_16x16x32_bf16(false, a, false, b, (short)0, c, false, false);
}

__device__ inline float hsig(float x) { return fminf(fmaxf(fmaf(x, 0.2f, 0.5f), 0.f), 1.f); }

__device__ inline float wred_sum(float v) {
#pragma unroll
  for (int m = 16; m > 0; m >>= 1) v += __shfl_xor(v, m, 32);
  return v;
}
__device__ inline float wred_max(float v) {
#pragma unroll
  for (int m = 16; m > 0; m >>= 1) v = fmaxf(v, __shfl_xor(v, m, 32));
  return v;
}

// One LSTM step for the d-slice owned by this wave. Valid on lanes 0..15.
__device__ inline void lstm_gates_wmma(const unsigned* xh_dw, const unsigned* __restrict__ wz,
                                       float bi, float bf_, float bg, float bo,
                                       int wave, int lane, float& c, float& h) {
  f32x8 a0 = {0.f,0.f,0.f,0.f,0.f,0.f,0.f,0.f};
  f32x8 a1 = a0, a2 = a0, a3 = a0;
#pragma unroll 4
  for (int kt = 0; kt < KT_; ++kt) {
    bf16x16 a  = load_a_frag(xh_dw, kt, lane);
    bf16x16 b0 = load_b_frag(wz, ( 0 + wave) * KT_ + kt, lane);
    bf16x16 b1 = load_b_frag(wz, (16 + wave) * KT_ + kt, lane);
    bf16x16 b2 = load_b_frag(wz, (32 + wave) * KT_ + kt, lane);
    bf16x16 b3 = load_b_frag(wz, (48 + wave) * KT_ + kt, lane);
    a0 = wmma_bf16(a, b0, a0);
    a1 = wmma_bf16(a, b1, a1);
    a2 = wmma_bf16(a, b2, a2);
    a3 = wmma_bf16(a, b3, a3);
  }
  float vi = hsig(a0[0] + bi);
  float vf = hsig(a1[0] + bf_);
  float vg = tanhf(a2[0] + bg);
  float vo = hsig(a3[0] + bo);
  c = vf * c + vi * vg;
  h = vo * tanhf(c);
}

__device__ inline float composer_wmma(const unsigned* om_dw, const unsigned* __restrict__ wc,
                                      float bias, int wave, int lane) {
  f32x8 acc = {0.f,0.f,0.f,0.f,0.f,0.f,0.f,0.f};
#pragma unroll 4
  for (int kt = 0; kt < KT_; ++kt) {
    bf16x16 a  = load_a_frag(om_dw, kt, lane);
    bf16x16 bm = load_b_frag(wc, wave * KT_ + kt, lane);
    acc = wmma_bf16(a, bm, acc);
  }
  return acc[0] + bias;
}

// ---------------- prep kernels ----------------

// Pack [W;U] (512 x 1024 fp32) -> bf16 WMMA B-fragment tiles.
__global__ void pack_wz(unsigned* __restrict__ dst,
                        const float* __restrict__ W, const float* __restrict__ U) {
  int g = blockIdx.x * blockDim.x + threadIdx.x;
  if (g >= 64 * KT_ * 256) return;
  int T = g >> 8, r = g & 255;
  int lane = r >> 3, j = r & 7;
  int half = lane >> 4;
  int ktile = T & (KT_ - 1), ntile = T / KT_;
  int n  = (ntile << 4) | (lane & 15);
  int k0 = ktile * 32 + 8 * half + 16 * (j >> 2) + 2 * (j & 3);
  float v0 = (k0     < 256) ? W[k0 * D4_ + n]       : U[(k0 - 256) * D4_ + n];
  float v1 = (k0 + 1 < 256) ? W[(k0 + 1) * D4_ + n] : U[(k0 - 255) * D4_ + n];
  dst[g] = pack_bf16(v0, v1);
}

// Pack W_c (512 x 256 fp32) -> bf16 tiles.
__global__ void pack_wc(unsigned* __restrict__ dst, const float* __restrict__ Wc) {
  int g = blockIdx.x * blockDim.x + threadIdx.x;
  if (g >= 16 * KT_ * 256) return;
  int T = g >> 8, r = g & 255;
  int lane = r >> 3, j = r & 7;
  int half = lane >> 4;
  int ktile = T & (KT_ - 1), ntile = T / KT_;
  int n  = (ntile << 4) | (lane & 15);
  int k0 = ktile * 32 + 8 * half + 16 * (j >> 2) + 2 * (j & 3);
  dst[g] = pack_bf16(Wc[k0 * D_ + n], Wc[(k0 + 1) * D_ + n]);
}

// memT[b][d][l] = x[b][l][d]  (transposed so phase-2 passes are lane-coalesced over l)
__global__ void transpose_mem(float* __restrict__ memT, const float* __restrict__ x) {
  int g = blockIdx.x * blockDim.x + threadIdx.x;
  if (g >= B_ * D_ * L_) return;
  int l = g & (L_ - 1);
  int d = (g >> 9) & (D_ - 1);
  int b = g >> 17;
  memT[g] = x[((size_t)b * L_ + l) * D_ + d];
}

// ---------------- persistent per-batch kernel ----------------

__global__ __launch_bounds__(512)
void nse_persist(const float* __restrict__ x,
                 const float* __restrict__ b_r, const float* __restrict__ b_w,
                 const float* __restrict__ b_c,
                 const unsigned* __restrict__ wzr, const unsigned* __restrict__ wzw,
                 const unsigned* __restrict__ wc,
                 float* __restrict__ memT, float* __restrict__ oseq,
                 float* __restrict__ out) {
  const int b    = blockIdx.x;
  const int tid  = threadIdx.x;
  const int wave = tid >> 5;
  const int lane = tid & 31;
  const int dl   = (wave << 4) | (lane & 15);   // d owned by this (wave, lane%16)

  __shared__ __align__(16) unsigned xh[256];    // 512 bf16 in fragment order
  __shared__ __align__(16) float    zs[512];    // attention weights (l = tid)
  __shared__ __align__(16) float    mrt[256];   // m_rt / ct staging
  __shared__ __align__(16) float    hsh[256];   // h staging
  __shared__ __align__(16) float    obuf[256];  // o_t / o_{t+1} staging
  __shared__ float red[16];

  const float* xb  = x    + (size_t)b * L_ * D_;
  float*       osb = oseq + (size_t)b * L_ * D_;
  float*       mTb = memT + (size_t)b * D_ * L_;

  if (tid < 256) { xh[tid] = 0u; hsh[tid] = 0.f; }
  __syncthreads();

  const float bri = b_r[0 * D_ + dl], brf = b_r[1 * D_ + dl];
  const float brg = b_r[2 * D_ + dl], bro = b_r[3 * D_ + dl];

  // ===== Phase 1: reader LSTM over L steps, emit o_seq =====
  float c_r = 0.f, h_r = 0.f;
  for (int t = 0; t < L_; ++t) {
    if (tid < 128) {
      const float2* xr = (const float2*)(xb + (size_t)t * D_);
      float2 v = xr[tid];
      xh[swz(tid)] = pack_bf16(v.x, v.y);
    }
    __syncthreads();
    lstm_gates_wmma(xh, wzr, bri, brf, brg, bro, wave, lane, c_r, h_r);
    if (lane < 16) {
      hsh[dl] = h_r;
      osb[(size_t)t * D_ + dl] = h_r;
    }
    __syncthreads();
    if (tid < 128) {
      float2 v = ((const float2*)hsh)[tid];
      xh[swz(128 + tid)] = pack_bf16(v.x, v.y);
    }
    __syncthreads();
  }

  // ===== Phase 2: compose-and-write loop =====
  if (tid < 256) { xh[tid] = 0u; hsh[tid] = 0.f; }
  __syncthreads();

  const float bwi = b_w[0 * D_ + dl], bwf = b_w[1 * D_ + dl];
  const float bwg = b_w[2 * D_ + dl], bwo = b_w[3 * D_ + dl];
  const float bcc = b_c[dl];

  float c_w = 0.f, h_w = 0.f;

  // initial scores: sc(l=tid) = o_0 . mem_0[:, l]
  if (tid < 256) obuf[tid] = osb[tid];
  __syncthreads();
  float sc = 0.f;
  {
    const float* mcol = mTb + tid;
#pragma unroll 4
    for (int d = 0; d < D_; ++d) sc = fmaf(obuf[d], mcol[(size_t)d * L_], sc);
  }

  for (int t = 0; t < L_; ++t) {
    __syncthreads();
    if (tid < 256) obuf[tid] = osb[(size_t)t * D_ + tid];   // o_t for composer

    // softmax over l (one score per thread)
    float m = wred_max(sc);
    if (lane == 0) red[wave] = m;
    __syncthreads();
    float mx = red[0];
#pragma unroll
    for (int i = 1; i < 16; ++i) mx = fmaxf(mx, red[i]);
    float e = __expf(sc - mx);
    float s = wred_sum(e);
    __syncthreads();
    if (lane == 0) red[wave] = s;
    __syncthreads();
    float ssum = 0.f;
#pragma unroll
    for (int i = 0; i < 16; ++i) ssum += red[i];
    float zl = e / ssum;
    zs[tid] = zl;
    __syncthreads();

    // m_rt[d] = sum_l z[l] * memT[d][l]
    if (tid < 256) {
      const float4* mr = (const float4*)(mTb + (size_t)tid * L_);
      const float4* zz = (const float4*)zs;
      float acc = 0.f;
#pragma unroll 4
      for (int i4 = 0; i4 < L_ / 4; ++i4) {
        float4 mv = mr[i4], zv = zz[i4];
        acc += mv.x * zv.x + mv.y * zv.y + mv.z * zv.z + mv.w * zv.w;
      }
      mrt[tid] = acc;
    }
    __syncthreads();

    // build om = [o_t | m_rt] in bf16 (fragment order)
    if (tid < 128) {
      float2 v = ((const float2*)obuf)[tid];
      xh[swz(tid)] = pack_bf16(v.x, v.y);
    } else if (tid < 256) {
      int u = tid - 128;
      float2 v = ((const float2*)mrt)[u];
      xh[swz(128 + u)] = pack_bf16(v.x, v.y);
    }
    __syncthreads();

    // composer: ct = om @ W_c + b_c
    float ct = composer_wmma(xh, wc, bcc, wave, lane);
    __syncthreads();
    if (lane < 16) mrt[dl] = ct;        // reuse mrt as ct staging
    __syncthreads();
    if (tid < 128) {
      float2 vc = ((const float2*)mrt)[tid];
      float2 vh = ((const float2*)hsh)[tid];
      xh[swz(tid)]       = pack_bf16(vc.x, vc.y);
      xh[swz(128 + tid)] = pack_bf16(vh.x, vh.y);
    }
    __syncthreads();

    // writer LSTM
    lstm_gates_wmma(xh, wzw, bwi, bwf, bwg, bwo, wave, lane, c_w, h_w);
    if (lane < 16) hsh[dl] = h_w;
    if (tid < 256) obuf[tid] = (t + 1 < L_) ? osb[(size_t)(t + 1) * D_ + tid] : 0.f;
    __syncthreads();

    // fused: mem update + next-step attention scores
    {
      float zme = zs[tid];
      float one_m = 1.f - zme;
      float nsc = 0.f;
      float* mcol = mTb + tid;
#pragma unroll 4
      for (int d = 0; d < D_; ++d) {
        float mv = mcol[(size_t)d * L_];
        mv = mv * one_m + hsh[d] * zme;
        mcol[(size_t)d * L_] = mv;
        nsc = fmaf(obuf[d], mv, nsc);
      }
      sc = nsc;
    }
  }

  if (lane < 16) out[b * D_ + dl] = h_w;
}

// ---------------- launch ----------------

extern "C" void kernel_launch(void* const* d_in, const int* in_sizes, int n_in,
                              void* d_out, int out_size, void* d_ws, size_t ws_size,
                              hipStream_t stream) {
  const float* x   = (const float*)d_in[0];
  const float* W_r = (const float*)d_in[1];
  const float* U_r = (const float*)d_in[2];
  const float* b_r = (const float*)d_in[3];
  const float* W_w = (const float*)d_in[4];
  const float* U_w = (const float*)d_in[5];
  const float* b_w = (const float*)d_in[6];
  const float* W_c = (const float*)d_in[7];
  const float* b_c = (const float*)d_in[8];

  char* ws = (char*)d_ws;
  unsigned* wzr  = (unsigned*)(ws);                                   // 1 MB
  unsigned* wzw  = (unsigned*)(ws + (1u << 20));                      // 1 MB
  unsigned* wc   = (unsigned*)(ws + (2u << 20));                      // 256 KB
  float*    memT = (float*)   (ws + (2u << 20) + (256u << 10));       // 4 MB
  float*    oseq = (float*)   (ws + (6u << 20) + (256u << 10));       // 4 MB

  pack_wz<<<262144 / 256, 256, 0, stream>>>(wzr, W_r, U_r);
  pack_wz<<<262144 / 256, 256, 0, stream>>>(wzw, W_w, U_w);
  pack_wc<<<65536 / 256, 256, 0, stream>>>(wc, W_c);
  transpose_mem<<<(B_ * D_ * L_) / 256, 256, 0, stream>>>(memT, x);
  nse_persist<<<B_, 512, 0, stream>>>(x, b_r, b_w, b_c, wzr, wzw, wc, memT, oseq,
                                      (float*)d_out);
}
